// CausalSelfAttention_27307402068092
// MI455X (gfx1250) — matrix operand order repaired and checked by reading, not compiled, not run on hardware
//
#include <hip/hip_runtime.h>
#include <hip/hip_bf16.h>

#define BSZ 4
#define SEQ 2048
#define DIM 768
#define NH  6
#define HD  128

typedef __bf16 bf16;
typedef __attribute__((ext_vector_type(16))) __bf16 v16bf;
typedef __attribute__((ext_vector_type(8)))  __bf16 v8bf;
typedef __attribute__((ext_vector_type(8)))  float  v8f;

#define WMMA_BF16(a, b, c) \
  __builtin_amdgcn_wmma_f32_16x16x32_bf16(false, (a), false, (b), (short)0, (c), false, false)

// A fragment: 16x32 bf16, row-major source. lane&15 = M row; lane half selects
// K halves {0..7,16..23} (lo lanes) / {8..15,24..31} (hi lanes).
static __device__ __forceinline__ v16bf load_fragA(const bf16* base, int stride, int lane) {
  const int m  = lane & 15;
  const int hi = lane >> 4;
  const bf16* p = base + (size_t)m * stride + hi * 8;
  v8bf lo = *(const v8bf*)(p);
  v8bf h8 = *(const v8bf*)(p + 16);
  v16bf r;
#pragma unroll
  for (int i = 0; i < 8; ++i) { r[i] = lo[i]; r[i + 8] = h8[i]; }
  return r;
}

// B fragment: 32x16 bf16, K-major source (row = K). lane&15 = N column; lane half
// selects K {0..7,16..23} (lo lanes) / {8..15,24..31} (hi lanes).
static __device__ __forceinline__ v16bf load_fragB(const bf16* base, int stride, int lane) {
  const int n  = lane & 15;
  const int hi = lane >> 4;
  const bf16* p = base + n + (size_t)(hi * 8) * stride;
  v16bf r;
#pragma unroll
  for (int i = 0; i < 8; ++i) { r[i] = p[(size_t)i * stride]; r[i + 8] = p[(size_t)(16 + i) * stride]; }
  return r;
}

__global__ void __launch_bounds__(256) cvt_f32_bf16(const float* __restrict__ in,
                                                    bf16* __restrict__ out, int n) {
  int i = blockIdx.x * blockDim.x + threadIdx.x;
  if (i < n) out[i] = (bf16)in[i];
}

// Fused QKV projection, 64x64 register-blocked tile per wave (16 WMMAs / k-step,
// ~33 FLOP/byte from L2). Q gets 1/sqrt(hd) folded in; K stored transposed
// [b,h,hd,L]; V stored [b,h,l,hd].
__global__ void __launch_bounds__(256) qkv_gemm(
    const bf16* __restrict__ xb,
    const bf16* __restrict__ wq, const bf16* __restrict__ wk, const bf16* __restrict__ wv,
    const float* __restrict__ bq, const float* __restrict__ bk, const float* __restrict__ bv,
    bf16* __restrict__ Q, bf16* __restrict__ Kt, bf16* __restrict__ V) {
  const int lane = threadIdx.x & 31;
  const int wave = (blockIdx.x * blockDim.x + threadIdx.x) >> 5;
  const int NT_N = DIM / 64;            // 12
  const int NT_M = (BSZ * SEQ) / 64;    // 128
  const int per  = NT_M * NT_N;         // 1536
  if (wave >= 3 * per) return;
  const int which = wave / per;
  const int r     = wave % per;
  const int mt    = r / NT_N;
  const int nt    = r % NT_N;
  const bf16*  w    = which == 0 ? wq : which == 1 ? wk : wv;
  const float* bias = which == 0 ? bq : which == 1 ? bk : bv;

  const int row0 = mt * 64, col0 = nt * 64;
  v8f acc[4][4] = {};
#pragma unroll 2
  for (int kk = 0; kk < DIM; kk += 32) {
    __builtin_prefetch(xb + (size_t)row0 * DIM + kk + 32, 0, 3);
    v16bf a[4], bfr[4];
#pragma unroll
    for (int j = 0; j < 4; ++j)
      bfr[j] = load_fragB(w + (size_t)kk * DIM + col0 + j * 16, DIM, lane);
#pragma unroll
    for (int i = 0; i < 4; ++i)
      a[i] = load_fragA(xb + (size_t)(row0 + i * 16) * DIM + kk, DIM, lane);
#pragma unroll
    for (int i = 0; i < 4; ++i)
#pragma unroll
      for (int j = 0; j < 4; ++j)
        acc[i][j] = WMMA_BF16(a[i], bfr[j], acc[i][j]);
  }
  const int n = lane & 15, half = lane >> 4;
  const float qscale = 0.08838834764831845f;  // 1/sqrt(128)
#pragma unroll
  for (int j = 0; j < 4; ++j) {
    const int col = col0 + j * 16 + n;
    const int h = col >> 7, hd = col & 127;
    const float bia = bias[col];
#pragma unroll
    for (int i = 0; i < 4; ++i)
#pragma unroll
      for (int v = 0; v < 8; ++v) {
        int row = row0 + i * 16 + half * 8 + v;  // flat token over BSZ*SEQ
        int b = row >> 11, l = row & (SEQ - 1);
        size_t bh = (size_t)(b * NH + h);
        float val = acc[i][j][v] + bia;
        if (which == 0)      Q[(bh * SEQ + l) * HD + hd]  = (bf16)(val * qscale);
        else if (which == 1) Kt[(bh * HD + hd) * SEQ + l] = (bf16)val;
        else                 V[(bh * SEQ + l) * HD + hd]  = (bf16)val;
      }
  }
}

// Flash-attention forward: one wave handles (b,h, 32-query pair), streaming
// 32-key tiles. Both 16-query tiles align with the key tile so every K/V
// fragment is reused twice; only the diagonal tile needs masking.
__global__ void __launch_bounds__(256) attn_fwd(
    const bf16* __restrict__ Q, const bf16* __restrict__ Kt,
    const bf16* __restrict__ V, bf16* __restrict__ AO) {
  __shared__ __align__(16) bf16 pshare[8][2 * 16 * 40];  // per-wave P tiles (padded)
  const int lane = threadIdx.x & 31;
  const int wib  = threadIdx.x >> 5;
  const int wave = (blockIdx.x * blockDim.x + threadIdx.x) >> 5;
  const int NQP  = SEQ / 32;  // 64 query pairs
  if (wave >= BSZ * NH * NQP) return;
  const int qp = wave % NQP;
  const int bh = wave / NQP;
  const int b = bh / NH, h = bh % NH;
  const int q0 = qp * 32;
  const bf16* Qb = Q  + (size_t)bh * SEQ * HD;
  const bf16* Kb = Kt + (size_t)bh * HD * SEQ;
  const bf16* Vb = V  + (size_t)bh * SEQ * HD;
  bf16* pbuf = &pshare[wib][0];

  v16bf qf[2][4];
#pragma unroll
  for (int t = 0; t < 2; ++t)
#pragma unroll
    for (int c = 0; c < 4; ++c)
      qf[t][c] = load_fragA(Qb + (size_t)(q0 + t * 16) * HD + c * 32, HD, lane);

  float mst[2][8], lst[2][8];
  v8f o[2][8] = {};
#pragma unroll
  for (int t = 0; t < 2; ++t)
#pragma unroll
    for (int v = 0; v < 8; ++v) { mst[t][v] = -3.0e38f; lst[t][v] = 0.f; }

  const int n = lane & 15, half = lane >> 4;
  const int nk = qp + 1;  // causal: key tiles 0..qp
  for (int j = 0; j < nk; ++j) {
    const int kb = j * 32;
    // scores: S[t][16q x 32k], K fragments shared by both q tiles
    v8f sacc[2][2] = {};
#pragma unroll
    for (int si = 0; si < 2; ++si)
#pragma unroll
      for (int c = 0; c < 4; ++c) {
        v16bf kf = load_fragB(Kb + (size_t)(c * 32) * SEQ + kb + si * 16, SEQ, lane);
        sacc[0][si] = WMMA_BF16(qf[0][c], kf, sacc[0][si]);
        sacc[1][si] = WMMA_BF16(qf[1][c], kf, sacc[1][si]);
      }
    if (j == nk - 1) {  // diagonal tile: causal mask for both q tiles
#pragma unroll
      for (int t = 0; t < 2; ++t)
#pragma unroll
        for (int si = 0; si < 2; ++si)
#pragma unroll
          for (int v = 0; v < 8; ++v) {
            int kg = kb + si * 16 + n;
            int qg = q0 + t * 16 + half * 8 + v;
            if (kg > qg) sacc[t][si][v] = -3.0e38f;
          }
    }
    // online softmax per q tile; row stats at m = vgpr + 8*half
#pragma unroll
    for (int t = 0; t < 2; ++t) {
      float mnew[8], alpha[8];
#pragma unroll
      for (int v = 0; v < 8; ++v) {
        float mx = fmaxf(sacc[t][0][v], sacc[t][1][v]);
        mx = fmaxf(mx, __shfl_xor(mx, 1, 32));
        mx = fmaxf(mx, __shfl_xor(mx, 2, 32));
        mx = fmaxf(mx, __shfl_xor(mx, 4, 32));
        mx = fmaxf(mx, __shfl_xor(mx, 8, 32));
        mnew[v]   = fmaxf(mst[t][v], mx);
        alpha[v]  = __expf(mst[t][v] - mnew[v]);
        mst[t][v] = mnew[v];
      }
#pragma unroll
      for (int v = 0; v < 8; ++v) {
        float e0 = __expf(sacc[t][0][v] - mnew[v]);
        float e1 = __expf(sacc[t][1][v] - mnew[v]);
        sacc[t][0][v] = e0; sacc[t][1][v] = e1;
        float rs = e0 + e1;
        rs += __shfl_xor(rs, 1, 32);
        rs += __shfl_xor(rs, 2, 32);
        rs += __shfl_xor(rs, 4, 32);
        rs += __shfl_xor(rs, 8, 32);
        lst[t][v] = lst[t][v] * alpha[v] + rs;
      }
#pragma unroll
      for (int f = 0; f < 8; ++f)
#pragma unroll
        for (int v = 0; v < 8; ++v) o[t][f][v] *= alpha[v];
      // P_t: D-fragment layout -> LDS [m][k]
#pragma unroll
      for (int si = 0; si < 2; ++si)
#pragma unroll
        for (int v = 0; v < 8; ++v)
          pbuf[t * 640 + (half * 8 + v) * 40 + si * 16 + n] = (bf16)sacc[t][si][v];
    }
    asm volatile("s_wait_dscnt 0" ::: "memory");
    v16bf pf0 = load_fragA(pbuf, 40, lane);
    v16bf pf1 = load_fragA(pbuf + 640, 40, lane);
    // PV: V fragments shared by both q tiles
#pragma unroll
    for (int f = 0; f < 8; ++f) {
      v16bf vf = load_fragB(Vb + (size_t)kb * HD + f * 16, HD, lane);
      o[0][f] = WMMA_BF16(pf0, vf, o[0][f]);
      o[1][f] = WMMA_BF16(pf1, vf, o[1][f]);
    }
  }
  // normalize and store attention output [B, L, D] as bf16
#pragma unroll
  for (int t = 0; t < 2; ++t)
#pragma unroll
    for (int f = 0; f < 8; ++f)
#pragma unroll
      for (int v = 0; v < 8; ++v) {
        int m = t * 16 + half * 8 + v;
        float val = o[t][f][v] / lst[t][v];
        AO[(size_t)(b * SEQ + q0 + m) * DIM + h * HD + f * 16 + n] = (bf16)val;
      }
}

// Output projection: out = AO @ Wo + bo, fp32 result. 64x64 tile per wave.
__global__ void __launch_bounds__(256) out_proj(
    const bf16* __restrict__ AO, const bf16* __restrict__ wo,
    const float* __restrict__ bo, float* __restrict__ out) {
  const int lane = threadIdx.x & 31;
  const int wave = (blockIdx.x * blockDim.x + threadIdx.x) >> 5;
  const int NT_N = DIM / 64;           // 12
  const int NT_M = (BSZ * SEQ) / 64;   // 128
  if (wave >= NT_M * NT_N) return;
  const int mt = wave / NT_N, nt = wave % NT_N;
  const int row0 = mt * 64, col0 = nt * 64;
  v8f acc[4][4] = {};
#pragma unroll 2
  for (int kk = 0; kk < DIM; kk += 32) {
    __builtin_prefetch(AO + (size_t)row0 * DIM + kk + 32, 0, 3);
    v16bf a[4], bfr[4];
#pragma unroll
    for (int j = 0; j < 4; ++j)
      bfr[j] = load_fragB(wo + (size_t)kk * DIM + col0 + j * 16, DIM, lane);
#pragma unroll
    for (int i = 0; i < 4; ++i)
      a[i] = load_fragA(AO + (size_t)(row0 + i * 16) * DIM + kk, DIM, lane);
#pragma unroll
    for (int i = 0; i < 4; ++i)
#pragma unroll
      for (int j = 0; j < 4; ++j)
        acc[i][j] = WMMA_BF16(a[i], bfr[j], acc[i][j]);
  }
  const int n = lane & 15, half = lane >> 4;
#pragma unroll
  for (int j = 0; j < 4; ++j) {
    const float bia = bo[col0 + j * 16 + n];
#pragma unroll
    for (int i = 0; i < 4; ++i)
#pragma unroll
      for (int v = 0; v < 8; ++v) {
        int row = row0 + i * 16 + half * 8 + v;
        out[(size_t)row * DIM + col0 + j * 16 + n] = acc[i][j][v] + bia;
      }
  }
}

extern "C" void kernel_launch(void* const* d_in, const int* in_sizes, int n_in,
                              void* d_out, int out_size, void* d_ws, size_t ws_size,
                              hipStream_t stream) {
  (void)in_sizes; (void)n_in; (void)out_size; (void)ws_size;
  const float* x  = (const float*)d_in[0];
  const float* Wq = (const float*)d_in[1];
  const float* bq = (const float*)d_in[2];
  const float* Wk = (const float*)d_in[3];
  const float* bk = (const float*)d_in[4];
  const float* Wv = (const float*)d_in[5];
  const float* bv = (const float*)d_in[6];
  const float* Wo = (const float*)d_in[7];
  const float* bo = (const float*)d_in[8];
  float* out = (float*)d_out;

  const size_t NX = (size_t)BSZ * SEQ * DIM;  // 6,291,456
  const size_t NW = (size_t)DIM * DIM;        //   589,824
  char* ws = (char*)d_ws;
  bf16* xb  = (bf16*)ws; ws += NX * 2;
  bf16* wqb = (bf16*)ws; ws += NW * 2;
  bf16* wkb = (bf16*)ws; ws += NW * 2;
  bf16* wvb = (bf16*)ws; ws += NW * 2;
  bf16* wob = (bf16*)ws; ws += NW * 2;
  bf16* Qd  = (bf16*)ws; ws += NX * 2;
  bf16* Ktd = (bf16*)ws; ws += NX * 2;
  bf16* Vd  = (bf16*)ws; ws += NX * 2;
  bf16* AOd = (bf16*)ws; ws += NX * 2;  // total ~64.5 MB of d_ws

  cvt_f32_bf16<<<(int)((NX + 255) / 256), 256, 0, stream>>>(x,  xb,  (int)NX);
  cvt_f32_bf16<<<(int)((NW + 255) / 256), 256, 0, stream>>>(Wq, wqb, (int)NW);
  cvt_f32_bf16<<<(int)((NW + 255) / 256), 256, 0, stream>>>(Wk, wkb, (int)NW);
  cvt_f32_bf16<<<(int)((NW + 255) / 256), 256, 0, stream>>>(Wv, wvb, (int)NW);
  cvt_f32_bf16<<<(int)((NW + 255) / 256), 256, 0, stream>>>(Wo, wob, (int)NW);

  // 3 * 128 * 12 = 4608 waves, 8 waves/block
  qkv_gemm<<<4608 / 8, 256, 0, stream>>>(xb, wqb, wkb, wvb, bq, bk, bv, Qd, Ktd, Vd);
  // 4 * 6 * 64 = 1536 waves
  attn_fwd<<<1536 / 8, 256, 0, stream>>>(Qd, Ktd, Vd, AOd);
  // 128 * 12 = 1536 waves
  out_proj<<<1536 / 8, 256, 0, stream>>>(AOd, wob, bo, out);
}
